// MoEProjectionLayer_26319559590549
// MI455X (gfx1250) — matrix operand order repaired and checked by reading, not compiled, not run on hardware
//
#include <hip/hip_runtime.h>
#include <hip/hip_bf16.h>
#include <math.h>

typedef __bf16 bf16_t;
typedef __attribute__((ext_vector_type(16))) __bf16 v16bf;
typedef __attribute__((ext_vector_type(8)))  float  v8f;

#define Bn 16384
#define Dn 1024
#define Hn 4096
#define En 8

#define TM 128
#define TN 256
#define TK 32
#define LDP (TK + 8)            // LDS pitch (bf16 elems): 80 B rows, 16B-aligned
#define ASTAGE (TM * LDP * 2)   // bytes per A stage buffer
#define BSTAGE (TN * LDP * 2)   // bytes per B stage buffer

// ------------------------------------------------ CDNA5 async global->LDS
__device__ __forceinline__ void async_copy_b128(unsigned int lds_off, const void* gptr) {
    asm volatile("global_load_async_to_lds_b128 %0, %1, off"
                 :: "v"(lds_off), "v"((unsigned long long)gptr)
                 : "memory");
}
__device__ __forceinline__ void wait_async0() {
    asm volatile("s_wait_asynccnt 0x0" ::: "memory");
}

// ---------------------------------------------------------------- utilities
__global__ __launch_bounds__(32)
void init_meta_kernel(int* counts, int* cursors) {
    if (threadIdx.x < En) { counts[threadIdx.x] = 0; cursors[threadIdx.x] = 0; }
}

__global__ __launch_bounds__(256)
void cvt_bf16_kernel(const float* __restrict__ src, bf16_t* __restrict__ dst, size_t n) {
    size_t stride = (size_t)gridDim.x * blockDim.x;
    for (size_t i = (size_t)blockIdx.x * blockDim.x + threadIdx.x; i < n; i += stride)
        dst[i] = (bf16_t)src[i];
}

// ---------------------------------------------------------------- gating
// one wave (32 lanes) per token: 8 expert logits, softmax->top2->renorm
__global__ __launch_bounds__(256)
void gate_kernel(const float* __restrict__ x, const float* __restrict__ gw,
                 const float* __restrict__ gb, int* __restrict__ ti,
                 float* __restrict__ tw, int* __restrict__ counts) {
    __shared__ float gl[En][Dn];                       // 32 KB
    for (int i = threadIdx.x; i < En * Dn; i += 256)
        ((float*)gl)[i] = gw[i];
    __syncthreads();

    const int lane = threadIdx.x & 31;
    const int wave = threadIdx.x >> 5;
    const int t = blockIdx.x * 8 + wave;
    const float* xr = x + (size_t)t * Dn;

    float acc[En];
#pragma unroll
    for (int e = 0; e < En; e++) acc[e] = 0.f;
    for (int it = 0; it < Dn / 32; it++) {
        const int d = lane + it * 32;
        const float xv = xr[d];
#pragma unroll
        for (int e = 0; e < En; e++) acc[e] = fmaf(xv, gl[e][d], acc[e]);
    }
#pragma unroll
    for (int e = 0; e < En; e++) {
        float v = acc[e];
        for (int off = 16; off; off >>= 1) v += __shfl_xor(v, off, 32);
        acc[e] = v + gb[e];
    }
    // softmax + top-2 + renorm: global denominator cancels, use exp(l - max)
    float mx = acc[0];
#pragma unroll
    for (int e = 1; e < En; e++) mx = fmaxf(mx, acc[e]);
    float p[En];
#pragma unroll
    for (int e = 0; e < En; e++) p[e] = __expf(acc[e] - mx);
    int i0 = 0;
#pragma unroll
    for (int e = 1; e < En; e++) if (p[e] > p[i0]) i0 = e;
    int i1 = (i0 == 0) ? 1 : 0;
#pragma unroll
    for (int e = 0; e < En; e++) if (e != i0 && p[e] > p[i1]) i1 = e;
    const float inv = 1.0f / (p[i0] + p[i1]);
    if (lane == 0) {
        ti[t * 2 + 0] = i0;  tw[t * 2 + 0] = p[i0] * inv;
        ti[t * 2 + 1] = i1;  tw[t * 2 + 1] = p[i1] * inv;
        atomicAdd(&counts[i0], 1);
        atomicAdd(&counts[i1], 1);
    }
}

__global__ __launch_bounds__(32)
void offsets_kernel(const int* __restrict__ counts, int* __restrict__ offsets) {
    if (threadIdx.x == 0) {
        int o = 0;
        for (int e = 0; e < En; e++) { offsets[e] = o; o += counts[e]; }
        offsets[En] = o;
    }
}

__global__ __launch_bounds__(256)
void scatter_kernel(const int* __restrict__ ti, const int* __restrict__ offsets,
                    int* __restrict__ cursors, int* __restrict__ token_list,
                    int* __restrict__ slot_of) {
    const int t = blockIdx.x * 256 + threadIdx.x;
#pragma unroll
    for (int k = 0; k < 2; k++) {
        const int e = ti[t * 2 + k];
        const int pos = offsets[e] + atomicAdd(&cursors[e], 1);
        token_list[pos] = t;
        slot_of[t * 2 + k] = pos;
    }
}

// ---------------------------------------------------------------- WMMA fragment helpers
// A fragment (16x32 bf16, MxK): lanes 0-15 hold M=lane, K {0..7,16..23};
// lanes 16-31 hold M=lane-16, K {8..15,24..31}.
__device__ __forceinline__ v16bf load_a_frag(const bf16_t* row, int lane) {
    const int kb = (lane >> 4) * 8;
    v16bf a;
    *(uint4*)&a         = *(const uint4*)(row + kb);
    *(((uint4*)&a) + 1) = *(const uint4*)(row + kb + 16);
    return a;
}
// B fragment (32x16 bf16, KxN), B tile stored N-major in LDS: lane holds
// N=lane&15, K {0..7,16..23} (lanes 0-15) or {8..15,24..31} (lanes 16-31).
__device__ __forceinline__ v16bf load_b_frag(const bf16_t* row, int lane) {
    const int kb = (lane >> 4) * 8;
    v16bf b;
    *(uint4*)&b         = *(const uint4*)(row + kb);
    *(((uint4*)&b) + 1) = *(const uint4*)(row + kb + 16);
    return b;
}

// ---------------------------------------------------------------- GEMM1: h = gelu(x_gather @ w1^T + b1)
// 128x256 workgroup tile, 8 waves in 2x4 grid, 64x64 wave tile = 16 wmma/K-step,
// double-buffered LDS fed by async global->LDS copies, one barrier per K-step.
__global__ __launch_bounds__(256)
void gemm1_kernel(const bf16_t* __restrict__ xb,     // [B, D]
                  const bf16_t* __restrict__ w1b,    // [E, H, D]
                  const float*  __restrict__ b1,     // [E, H]
                  const int* __restrict__ token_list,
                  const int* __restrict__ counts,
                  const int* __restrict__ offsets,
                  bf16_t* __restrict__ h) {          // [2B, H]
    const int e = blockIdx.z;
    const int cnt = counts[e];
    const int m0 = blockIdx.y * TM;
    if (m0 >= cnt) return;                           // over-launched tile: early exit
    const int n0 = blockIdx.x * TN;
    const int off = offsets[e];

    __shared__ __align__(16) bf16_t Al[2][TM][LDP];  // 20 KB
    __shared__ __align__(16) bf16_t Bl[2][TN][LDP];  // 40 KB

    const int tid = threadIdx.x;
    const int lane = tid & 31, wave = tid >> 5;
    const int wm = (wave & 1) * 64;                  // 2 waves along M
    const int wn = (wave >> 1) * 64;                 // 4 waves along N

    // cooperative stage loads: A 128x32, B 256x32 (bf16)
    const int lrow = tid >> 1;
    const int lhalf = (tid & 1) * 16;
    const int am = m0 + lrow;
    const int slotIdx = off + (am < cnt ? am : cnt - 1);
    const bf16_t* aG  = xb + (size_t)token_list[slotIdx] * Dn + lhalf;
    const bf16_t* bG0 = w1b + (size_t)e * Hn * Dn + (size_t)(n0 + lrow) * Dn + lhalf;
    const bf16_t* bG1 = bG0 + (size_t)128 * Dn;

    const unsigned int aBase = (unsigned int)(size_t)&Al[0][0][0]
                             + (unsigned int)(lrow * (LDP * 2) + lhalf * 2);
    const unsigned int bBase = (unsigned int)(size_t)&Bl[0][0][0]
                             + (unsigned int)(lrow * (LDP * 2) + lhalf * 2);

    auto issue_stage = [&](int buf, int k0) {
        const unsigned int aL = aBase + (unsigned int)buf * ASTAGE;
        const unsigned int bL = bBase + (unsigned int)buf * BSTAGE;
        async_copy_b128(aL,      aG + k0);
        async_copy_b128(aL + 16, aG + k0 + 8);
        async_copy_b128(bL,      bG0 + k0);
        async_copy_b128(bL + 16, bG0 + k0 + 8);
        async_copy_b128(bL + 128 * (LDP * 2),      bG1 + k0);
        async_copy_b128(bL + 128 * (LDP * 2) + 16, bG1 + k0 + 8);
    };

    v8f acc[4][4];
    const v8f zero = {};
#pragma unroll
    for (int i = 0; i < 4; i++)
#pragma unroll
        for (int j = 0; j < 4; j++) acc[i][j] = zero;

    issue_stage(0, 0);
    wait_async0();
    __syncthreads();

    int cur = 0;
    for (int k0 = 0; k0 < Dn; k0 += TK) {
        if (k0 + TK < Dn) issue_stage(cur ^ 1, k0 + TK);

        v16bf af[4], bf[4];
#pragma unroll
        for (int i = 0; i < 4; i++)
            af[i] = load_a_frag(&Al[cur][wm + i * 16 + (lane & 15)][0], lane);
#pragma unroll
        for (int j = 0; j < 4; j++)
            bf[j] = load_b_frag(&Bl[cur][wn + j * 16 + (lane & 15)][0], lane);
#pragma unroll
        for (int i = 0; i < 4; i++)
#pragma unroll
            for (int j = 0; j < 4; j++)
                acc[i][j] = __builtin_amdgcn_wmma_f32_16x16x32_bf16(
                    false, af[i], false, bf[j], (short)0, acc[i][j], false, false);

        wait_async0();       // next stage resident in LDS
        __syncthreads();     // all waves done with stage `cur`, see stage `cur^1`
        cur ^= 1;
    }

    // epilogue: bias + exact gelu, store bf16 h
    const int lm = (lane >> 4) * 8;
    const int ln = lane & 15;
#pragma unroll
    for (int i = 0; i < 4; i++) {
#pragma unroll
        for (int j = 0; j < 4; j++) {
            const int col = n0 + wn + j * 16 + ln;
            const float bias = b1[e * Hn + col];
#pragma unroll
            for (int r = 0; r < 8; r++) {
                const int m = m0 + wm + i * 16 + lm + r;
                if (m < cnt) {
                    float v = acc[i][j][r] + bias;
                    v = 0.5f * v * (1.0f + erff(v * 0.70710678f));   // exact gelu
                    h[(size_t)(off + m) * Hn + col] = (bf16_t)v;
                }
            }
        }
    }
}

// ---------------------------------------------------------------- GEMM2: y = h @ w2^T + b2
__global__ __launch_bounds__(256)
void gemm2_kernel(const bf16_t* __restrict__ h,      // [2B, H]
                  const bf16_t* __restrict__ w2b,    // [E, D, H]
                  const float*  __restrict__ b2,     // [E, D]
                  const int* __restrict__ counts,
                  const int* __restrict__ offsets,
                  float* __restrict__ y) {           // [2B, D]
    const int e = blockIdx.z;
    const int cnt = counts[e];
    const int m0 = blockIdx.y * TM;
    if (m0 >= cnt) return;
    const int n0 = blockIdx.x * TN;
    const int off = offsets[e];

    __shared__ __align__(16) bf16_t Al[2][TM][LDP];
    __shared__ __align__(16) bf16_t Bl[2][TN][LDP];

    const int tid = threadIdx.x;
    const int lane = tid & 31, wave = tid >> 5;
    const int wm = (wave & 1) * 64;
    const int wn = (wave >> 1) * 64;

    const int lrow = tid >> 1;
    const int lhalf = (tid & 1) * 16;
    const int am = m0 + lrow;
    const int rowIdx = off + (am < cnt ? am : cnt - 1);
    const bf16_t* aG  = h + (size_t)rowIdx * Hn + lhalf;
    const bf16_t* bG0 = w2b + (size_t)e * Dn * Hn + (size_t)(n0 + lrow) * Hn + lhalf;
    const bf16_t* bG1 = bG0 + (size_t)128 * Hn;

    const unsigned int aBase = (unsigned int)(size_t)&Al[0][0][0]
                             + (unsigned int)(lrow * (LDP * 2) + lhalf * 2);
    const unsigned int bBase = (unsigned int)(size_t)&Bl[0][0][0]
                             + (unsigned int)(lrow * (LDP * 2) + lhalf * 2);

    auto issue_stage = [&](int buf, int k0) {
        const unsigned int aL = aBase + (unsigned int)buf * ASTAGE;
        const unsigned int bL = bBase + (unsigned int)buf * BSTAGE;
        async_copy_b128(aL,      aG + k0);
        async_copy_b128(aL + 16, aG + k0 + 8);
        async_copy_b128(bL,      bG0 + k0);
        async_copy_b128(bL + 16, bG0 + k0 + 8);
        async_copy_b128(bL + 128 * (LDP * 2),      bG1 + k0);
        async_copy_b128(bL + 128 * (LDP * 2) + 16, bG1 + k0 + 8);
    };

    v8f acc[4][4];
    const v8f zero = {};
#pragma unroll
    for (int i = 0; i < 4; i++)
#pragma unroll
        for (int j = 0; j < 4; j++) acc[i][j] = zero;

    issue_stage(0, 0);
    wait_async0();
    __syncthreads();

    int cur = 0;
    for (int k0 = 0; k0 < Hn; k0 += TK) {
        if (k0 + TK < Hn) issue_stage(cur ^ 1, k0 + TK);

        v16bf af[4], bf[4];
#pragma unroll
        for (int i = 0; i < 4; i++)
            af[i] = load_a_frag(&Al[cur][wm + i * 16 + (lane & 15)][0], lane);
#pragma unroll
        for (int j = 0; j < 4; j++)
            bf[j] = load_b_frag(&Bl[cur][wn + j * 16 + (lane & 15)][0], lane);
#pragma unroll
        for (int i = 0; i < 4; i++)
#pragma unroll
            for (int j = 0; j < 4; j++)
                acc[i][j] = __builtin_amdgcn_wmma_f32_16x16x32_bf16(
                    false, af[i], false, bf[j], (short)0, acc[i][j], false, false);

        wait_async0();
        __syncthreads();
        cur ^= 1;
    }

    const int lm = (lane >> 4) * 8;
    const int ln = lane & 15;
#pragma unroll
    for (int i = 0; i < 4; i++) {
#pragma unroll
        for (int j = 0; j < 4; j++) {
            const int col = n0 + wn + j * 16 + ln;
            const float bias = b2[e * Dn + col];
#pragma unroll
            for (int r = 0; r < 8; r++) {
                const int m = m0 + wm + i * 16 + lm + r;
                if (m < cnt)
                    y[(size_t)(off + m) * Dn + col] = acc[i][j][r] + bias;
            }
        }
    }
}

// ---------------------------------------------------------------- combine: per-token LN + weighted sum
__global__ __launch_bounds__(256)
void combine_kernel(const float* __restrict__ y, const int* __restrict__ ti,
                    const float* __restrict__ tw, const int* __restrict__ slot_of,
                    const float* __restrict__ lng, const float* __restrict__ lnb,
                    float* __restrict__ out) {
    const int t = blockIdx.x;
    __shared__ float red[256];
    float accv[Dn / 256];
#pragma unroll
    for (int c = 0; c < Dn / 256; c++) accv[c] = 0.f;

#pragma unroll
    for (int k = 0; k < 2; k++) {
        const int slot = slot_of[t * 2 + k];
        const int e = ti[t * 2 + k];
        const float w = tw[t * 2 + k];
        const float* yr = y + (size_t)slot * Dn;

        float s = 0.f, ss = 0.f;
        for (int i = threadIdx.x; i < Dn; i += 256) {
            const float v = yr[i];
            s += v; ss += v * v;
        }
        red[threadIdx.x] = s; __syncthreads();
        for (int st = 128; st > 0; st >>= 1) {
            if (threadIdx.x < st) red[threadIdx.x] += red[threadIdx.x + st];
            __syncthreads();
        }
        const float mean = red[0] / Dn;
        __syncthreads();
        red[threadIdx.x] = ss; __syncthreads();
        for (int st = 128; st > 0; st >>= 1) {
            if (threadIdx.x < st) red[threadIdx.x] += red[threadIdx.x + st];
            __syncthreads();
        }
        const float var = red[0] / Dn - mean * mean;
        const float rstd = rsqrtf(var + 1e-5f);
        __syncthreads();

        for (int i = threadIdx.x, c = 0; i < Dn; i += 256, c++) {
            const float v = (yr[i] - mean) * rstd * lng[(size_t)e * Dn + i] + lnb[(size_t)e * Dn + i];
            accv[c] += w * v;
        }
    }
    for (int i = threadIdx.x, c = 0; i < Dn; i += 256, c++)
        out[(size_t)t * Dn + i] = accv[c];
}

// ---------------------------------------------------------------- launch
extern "C" void kernel_launch(void* const* d_in, const int* in_sizes, int n_in,
                              void* d_out, int out_size, void* d_ws, size_t ws_size,
                              hipStream_t stream) {
    const float* x   = (const float*)d_in[0];
    const float* gw  = (const float*)d_in[1];
    const float* gb  = (const float*)d_in[2];
    const float* w1  = (const float*)d_in[3];
    const float* b1  = (const float*)d_in[4];
    const float* w2  = (const float*)d_in[5];
    const float* b2  = (const float*)d_in[6];
    const float* lng = (const float*)d_in[7];
    const float* lnb = (const float*)d_in[8];
    float* out = (float*)d_out;

    // workspace carve-up (~571 MB total)
    char* ws = (char*)d_ws;
    size_t o = 0;
    auto carve = [&](size_t bytes) -> void* {
        void* p = ws + o;
        o = (o + bytes + 255) & ~(size_t)255;
        return p;
    };
    bf16_t* xb     = (bf16_t*)carve((size_t)Bn * Dn * sizeof(bf16_t));       //  33.5 MB
    bf16_t* w1b    = (bf16_t*)carve((size_t)En * Hn * Dn * sizeof(bf16_t));  //  67.1 MB
    bf16_t* w2b    = (bf16_t*)carve((size_t)En * Hn * Dn * sizeof(bf16_t));  //  67.1 MB
    bf16_t* hbuf   = (bf16_t*)carve((size_t)2 * Bn * Hn * sizeof(bf16_t));   // 268.4 MB
    float*  ybuf   = (float*) carve((size_t)2 * Bn * Dn * sizeof(float));    // 134.2 MB
    float*  tw     = (float*) carve((size_t)Bn * 2 * sizeof(float));
    int*    ti     = (int*)   carve((size_t)Bn * 2 * sizeof(int));
    int*    slots  = (int*)   carve((size_t)Bn * 2 * sizeof(int));
    int*    tlist  = (int*)   carve((size_t)2 * Bn * sizeof(int));
    int*    counts = (int*)   carve(En * sizeof(int));
    int*    offs   = (int*)   carve((En + 1) * sizeof(int));
    int*    curs   = (int*)   carve(En * sizeof(int));

    init_meta_kernel<<<1, 32, 0, stream>>>(counts, curs);
    cvt_bf16_kernel<<<2048, 256, 0, stream>>>(x,  xb,  (size_t)Bn * Dn);
    cvt_bf16_kernel<<<4096, 256, 0, stream>>>(w1, w1b, (size_t)En * Hn * Dn);
    cvt_bf16_kernel<<<4096, 256, 0, stream>>>(w2, w2b, (size_t)En * Hn * Dn);
    gate_kernel<<<Bn / 8, 256, 0, stream>>>(x, gw, gb, ti, tw, counts);
    offsets_kernel<<<1, 32, 0, stream>>>(counts, offs);
    scatter_kernel<<<Bn / 256, 256, 0, stream>>>(ti, offs, curs, tlist, slots);
    gemm1_kernel<<<dim3(Hn / TN, Bn / TM, En), 256, 0, stream>>>(
        xb, w1b, b1, tlist, counts, offs, hbuf);
    gemm2_kernel<<<dim3(Dn / TN, Bn / TM, En), 256, 0, stream>>>(
        hbuf, w2b, b2, counts, offs, ybuf);
    combine_kernel<<<Bn, 256, 0, stream>>>(ybuf, ti, tw, slots, lng, lnb, out);
}
